// GINNet_NC_6837587935810
// MI455X (gfx1250) — matrix operand order repaired and checked by reading, not compiled, not run on hardware
//
#include <hip/hip_runtime.h>
#include <hip/hip_bf16.h>

typedef __attribute__((ext_vector_type(16))) __bf16 v16bf;
typedef __attribute__((ext_vector_type(8)))  float  v8f;

#define N_NODES 50000
#define N_EDGES 800000
#define DIM     512
#define NF      ((size_t)N_NODES * DIM)      // 25.6M floats
#define KN      (DIM * DIM)                  // 262144

// Pack two f32 into one dword of bf16 pairs (lowers to v_cvt_pk_bf16_f32).
__device__ inline unsigned pack_bf16(float a, float b) {
  __bf16 x = (__bf16)a, y = (__bf16)b;
  unsigned short ux = __builtin_bit_cast(unsigned short, x);
  unsigned short uy = __builtin_bit_cast(unsigned short, y);
  return (unsigned)ux | ((unsigned)uy << 16);
}

// ---------------------------------------------------------------------------
// Weights: f32 [k][n] -> bf16 TRANSPOSED [n][k] (once per launch).
// K-contiguous layout lets the GEMM stage B tiles with b128 copies only.
// ---------------------------------------------------------------------------
__global__ __launch_bounds__(256) void f32_to_bf16_t_kernel(
    const float* __restrict__ in, __bf16* __restrict__ out) {
  int t  = blockIdx.x * blockDim.x + threadIdx.x;   // 65536 threads
  int n  = t >> 7;                                  // 0..511
  int k4 = (t & 127) * 4;                           // 0..508
  float a = in[(size_t)(k4 + 0) * DIM + n];
  float b = in[(size_t)(k4 + 1) * DIM + n];
  float c = in[(size_t)(k4 + 2) * DIM + n];
  float d = in[(size_t)(k4 + 3) * DIM + n];
  uint2 o;
  o.x = pack_bf16(a, b);
  o.y = pack_bf16(c, d);
  *(uint2*)(out + (size_t)n * DIM + k4) = o;
}

// ---------------------------------------------------------------------------
// agg = (1 + eps) * x   (elementwise, float4)
// ---------------------------------------------------------------------------
__global__ __launch_bounds__(256) void init_scale_kernel(
    float* __restrict__ agg, const float* __restrict__ x,
    const float* __restrict__ eps, int n4) {
  int i = blockIdx.x * blockDim.x + threadIdx.x;
  if (i >= n4) return;
  float s = 1.0f + eps[0];
  float4 v = ((const float4*)x)[i];
  v.x *= s; v.y *= s; v.z *= s; v.w *= s;
  ((float4*)agg)[i] = v;
}

// ---------------------------------------------------------------------------
// agg[dst] += x[src] over all edges. One wave (32 lanes) per edge,
// 4 passes of float4 per lane covers 512 floats. HW f32 atomics into L2
// (feature matrix is L2-resident: 102 MB < 192 MB).
// ---------------------------------------------------------------------------
__global__ __launch_bounds__(256) void scatter_add_kernel(
    float* __restrict__ agg, const float* __restrict__ x,
    const int* __restrict__ ei, int E) {
  int e    = blockIdx.x * 8 + (threadIdx.x >> 5);
  int lane = threadIdx.x & 31;
  if (e >= E) return;
  int src = ei[e];
  int dst = ei[E + e];
  const float4* xs = (const float4*)(x + (size_t)src * DIM);
  float*        ad = agg + (size_t)dst * DIM;
#pragma unroll
  for (int p = 0; p < 4; ++p) {
    int c4 = p * 32 + lane;          // float4 index within the row
    float4 v = xs[c4];
    int cb = c4 * 4;
    unsafeAtomicAdd(ad + cb + 0, v.x);
    unsafeAtomicAdd(ad + cb + 1, v.y);
    unsafeAtomicAdd(ad + cb + 2, v.z);
    unsafeAtomicAdd(ad + cb + 3, v.w);
  }
}

// ---------------------------------------------------------------------------
// out = act(A @ W + b), A: [M,512] f32, Wt: [n][k] bf16, out: [M,512] f32.
// Block = 256 threads = 8 waves; block tile 128(M) x 64(N); each wave owns
// 16 rows and 4 N-subtiles of 16; K stepped by 32 via
// v_wmma_f32_16x16x32_bf16. All LDS staging is b128 (row stride 40 bf16 =
// 80B keeps 16B alignment). B fragments are batch-loaded so the 4 WMMAs
// issue back-to-back behind a single s_wait_dscnt.
// ---------------------------------------------------------------------------
__global__ __launch_bounds__(256) void gemm_bias_relu_kernel(
    const float*  __restrict__ A,
    const __bf16* __restrict__ Wt,
    const float*  __restrict__ bias,
    float*        __restrict__ out,
    int M, int apply_relu) {
  __shared__ __bf16 As[128 * 40];
  __shared__ __bf16 Bs[64 * 40];

  const int tid  = threadIdx.x;
  const int lane = tid & 31;
  const int wave = tid >> 5;
  const int m0   = blockIdx.y * 128;
  const int n0   = blockIdx.x * 64;

  v8f acc[4] = {};

  // ISA 16-bit A fragment mapping (wave32): lane&15 = row, lane>>4 selects
  // K halves {kb..kb+7, kb+16..kb+23}, kb = (lane>>4)*8.
  const int mloc = lane & 15;
  const int kb   = (lane >> 4) * 8;
  const int kb2  = (lane >> 4) * 16;   // unused: B fragment uses the same K order as A (kb)

  // A-tile staging role for this thread
  const int ar    = tid >> 1;
  const int ahalf = (tid & 1) * 16;
  const int agm   = m0 + ar;
  // B-tile staging role for this thread
  const int bn = tid >> 2;            // 0..63
  const int bk = (tid & 3) * 8;       // 0..24

  for (int k0 = 0; k0 < DIM; k0 += 32) {
    // ---- stage A tile: 128 rows x 32 K, f32 -> packed bf16, 2x b128 ----
    {
      unsigned pk[8];
      if (agm < M) {
        const float4* src = (const float4*)(A + (size_t)agm * DIM + k0 + ahalf);
#pragma unroll
        for (int q = 0; q < 4; ++q) {
          float4 v = src[q];
          pk[2 * q + 0] = pack_bf16(v.x, v.y);
          pk[2 * q + 1] = pack_bf16(v.z, v.w);
        }
        if (k0 + 32 < DIM)   // hint next K tile into cache (global_prefetch_b8)
          __builtin_prefetch(A + (size_t)agm * DIM + k0 + 32 + ahalf, 0, 1);
      } else {
#pragma unroll
        for (int q = 0; q < 8; ++q) pk[q] = 0u;
      }
      unsigned* dst = (unsigned*)&As[ar * 40 + ahalf];
      ((uint4*)dst)[0] = make_uint4(pk[0], pk[1], pk[2], pk[3]);
      ((uint4*)dst)[1] = make_uint4(pk[4], pk[5], pk[6], pk[7]);
    }
    // ---- stage W tile: straight b128 copy (pre-transposed bf16 [n][k]) ----
    {
      uint4 raw = *(const uint4*)(Wt + (size_t)(n0 + bn) * DIM + k0 + bk);
      *(uint4*)&Bs[bn * 40 + bk] = raw;
    }
    __syncthreads();

    // ---- fragments: A once, all 4 B fragments batched ------------------
    const __bf16* ap = &As[(wave * 16 + mloc) * 40 + kb];
    v16bf a;
#pragma unroll
    for (int j = 0; j < 8; ++j) { a[j] = ap[j]; a[8 + j] = ap[16 + j]; }

    v16bf b[4];
#pragma unroll
    for (int nt = 0; nt < 4; ++nt) {
      const __bf16* bp = &Bs[(nt * 16 + mloc) * 40 + kb];
#pragma unroll
      for (int j = 0; j < 8; ++j) { b[nt][j] = bp[j]; b[nt][8 + j] = bp[16 + j]; }
    }

    // ---- 4 WMMAs back-to-back (independent accumulators) ---------------
#pragma unroll
    for (int nt = 0; nt < 4; ++nt) {
      acc[nt] = __builtin_amdgcn_wmma_f32_16x16x32_bf16(
          false, a, false, b[nt], (short)0, acc[nt], false, false);
    }
    __syncthreads();
  }

  // ---- epilogue: bias + optional ReLU, f32 store ----------------------
  const int mhi  = (lane >> 4) * 8;   // C/D layout: VGPR v -> row v (+8 hi lanes)
  const int ncol = lane & 15;
#pragma unroll
  for (int nt = 0; nt < 4; ++nt) {
    int gn = n0 + nt * 16 + ncol;
    float bv = bias[gn];
#pragma unroll
    for (int v = 0; v < 8; ++v) {
      int gm = m0 + wave * 16 + v + mhi;
      if (gm < M) {
        float val = acc[nt][v] + bv;
        if (apply_relu) val = fmaxf(val, 0.0f);
        out[(size_t)gm * DIM + gn] = val;
      }
    }
  }
}

// ---------------------------------------------------------------------------
// Wave reductions (wave32)
// ---------------------------------------------------------------------------
__device__ inline float wave_sum(float s) {
#pragma unroll
  for (int off = 16; off > 0; off >>= 1) s += __shfl_xor(s, off, 32);
  return s;
}
__device__ inline float wave_max(float s) {
#pragma unroll
  for (int off = 16; off > 0; off >>= 1) s = fmaxf(s, __shfl_xor(s, off, 32));
  return s;
}

// Row L2 normalize (in place). relu(l2norm(relu(x))) == l2norm(relu(x))
// since the norm scale is positive, so this is the full inter-layer epilogue.
__global__ __launch_bounds__(256) void l2norm_rows_kernel(float* __restrict__ h, int M) {
  int row  = blockIdx.x * 8 + (threadIdx.x >> 5);
  int lane = threadIdx.x & 31;
  if (row >= M) return;
  float4* p = (float4*)(h + (size_t)row * DIM);
  float4 v[4];
  float s = 0.0f;
#pragma unroll
  for (int q = 0; q < 4; ++q) {
    v[q] = p[q * 32 + lane];
    s += v[q].x * v[q].x + v[q].y * v[q].y + v[q].z * v[q].z + v[q].w * v[q].w;
  }
  s = wave_sum(s);
  float inv = 1.0f / fmaxf(sqrtf(s), 1e-12f);
#pragma unroll
  for (int q = 0; q < 4; ++q) {
    v[q].x *= inv; v[q].y *= inv; v[q].z *= inv; v[q].w *= inv;
    p[q * 32 + lane] = v[q];
  }
}

// Softmax per row: probs = softmax(logits). One wave per row.
__global__ __launch_bounds__(256) void softmax_rows_kernel(
    const float* __restrict__ logits, float* __restrict__ probs, int M) {
  int row  = blockIdx.x * 8 + (threadIdx.x >> 5);
  int lane = threadIdx.x & 31;
  if (row >= M) return;
  const float4* p = (const float4*)(logits + (size_t)row * DIM);
  float4 v[4];
  float m = -3.402823e38f;
#pragma unroll
  for (int q = 0; q < 4; ++q) {
    v[q] = p[q * 32 + lane];
    m = fmaxf(m, fmaxf(fmaxf(v[q].x, v[q].y), fmaxf(v[q].z, v[q].w)));
  }
  m = wave_max(m);
  float s = 0.0f;
#pragma unroll
  for (int q = 0; q < 4; ++q) {
    v[q].x = __expf(v[q].x - m); v[q].y = __expf(v[q].y - m);
    v[q].z = __expf(v[q].z - m); v[q].w = __expf(v[q].w - m);
    s += v[q].x + v[q].y + v[q].z + v[q].w;
  }
  s = wave_sum(s);
  float inv = 1.0f / s;
  float4* o = (float4*)(probs + (size_t)row * DIM);
#pragma unroll
  for (int q = 0; q < 4; ++q) {
    v[q].x *= inv; v[q].y *= inv; v[q].z *= inv; v[q].w *= inv;
    o[q * 32 + lane] = v[q];
  }
}

// ---------------------------------------------------------------------------
extern "C" void kernel_launch(void* const* d_in, const int* in_sizes, int n_in,
                              void* d_out, int out_size, void* d_ws, size_t ws_size,
                              hipStream_t stream) {
  const float* x  = (const float*)d_in[0];
  const int*   ei = (const int*)d_in[1];
  const float* W1 = (const float*)d_in[2];
  const float* b1 = (const float*)d_in[3];
  const float* e1 = (const float*)d_in[4];
  const float* W2 = (const float*)d_in[5];
  const float* b2 = (const float*)d_in[6];
  const float* e2 = (const float*)d_in[7];
  const float* W3 = (const float*)d_in[8];
  const float* b3 = (const float*)d_in[9];
  const float* e3 = (const float*)d_in[10];

  float* out_logits = (float*)d_out;          // [50000, 512]
  float* out_probs  = out_logits + NF;        // [50000, 512]

  float*  agg = (float*)d_ws;                 // 102.4 MB
  float*  h   = agg + NF;                     // 102.4 MB
  __bf16* wb  = (__bf16*)(h + NF);            // 3 x 512KB bf16 weights (transposed)

  const dim3 gemm_grid(DIM / 64, (N_NODES + 127) / 128);  // (8, 391)
  const int  scatter_blocks = (N_EDGES + 7) / 8;          // 100000
  const int  elem4_blocks   = (int)(NF / 4 / 256);        // 25000
  const int  row_blocks     = (N_NODES + 7) / 8;          // 6250

  // Weights -> bf16, transposed to [n][k]
  f32_to_bf16_t_kernel<<<KN / 4 / 256, 256, 0, stream>>>(W1, wb);
  f32_to_bf16_t_kernel<<<KN / 4 / 256, 256, 0, stream>>>(W2, wb + KN);
  f32_to_bf16_t_kernel<<<KN / 4 / 256, 256, 0, stream>>>(W3, wb + 2 * KN);

  // ---- layer 1 ----
  init_scale_kernel<<<elem4_blocks, 256, 0, stream>>>(agg, x, e1, (int)(NF / 4));
  scatter_add_kernel<<<scatter_blocks, 256, 0, stream>>>(agg, x, ei, N_EDGES);
  gemm_bias_relu_kernel<<<gemm_grid, 256, 0, stream>>>(agg, wb, b1, h, N_NODES, 1);
  l2norm_rows_kernel<<<row_blocks, 256, 0, stream>>>(h, N_NODES);

  // ---- layer 2 ----
  init_scale_kernel<<<elem4_blocks, 256, 0, stream>>>(agg, h, e2, (int)(NF / 4));
  scatter_add_kernel<<<scatter_blocks, 256, 0, stream>>>(agg, h, ei, N_EDGES);
  gemm_bias_relu_kernel<<<gemm_grid, 256, 0, stream>>>(agg, wb + KN, b2, h, N_NODES, 1);
  l2norm_rows_kernel<<<row_blocks, 256, 0, stream>>>(h, N_NODES);

  // ---- layer 3 (reference's _gin_layer also applies ReLU -> logits) ----
  init_scale_kernel<<<elem4_blocks, 256, 0, stream>>>(agg, h, e3, (int)(NF / 4));
  scatter_add_kernel<<<scatter_blocks, 256, 0, stream>>>(agg, h, ei, N_EDGES);
  gemm_bias_relu_kernel<<<gemm_grid, 256, 0, stream>>>(agg, wb + 2 * KN, b3,
                                                       out_logits, N_NODES, 1);

  // ---- softmax ----
  softmax_rows_kernel<<<row_blocks, 256, 0, stream>>>(out_logits, out_probs, N_NODES);
}
